// TextRNN_attention_34248069218872
// MI455X (gfx1250) — matrix operand order repaired and checked by reading, not compiled, not run on hardware
//
#include <hip/hip_runtime.h>
#include <hip/hip_bf16.h>
#include <math.h>

// ---------------- problem constants ----------------
constexpr int N_CLASS = 32000;
constexpr int EMB     = 256;
constexpr int HID     = 512;
constexpr int BATCH   = 128;
constexpr int SEQ     = 512;
constexpr int CAT     = 2 * HID;   // 1024

// ---------------- WMMA types (gfx1250, wave32) ----------------
typedef __attribute__((ext_vector_type(16))) __bf16 v16bf;
typedef __attribute__((ext_vector_type(8)))  float  v8f;
typedef __attribute__((ext_vector_type(4)))  int    v4i_;

// A-fragment (16x32 bf16, row-major source): lane L -> row M = L%16, K-half = L/16.
// VGPR v holds K-pair at offset (v<4 ? 2v : 2v+8) + 8*(L/16).   [ISA 7.12.2]
__device__ __forceinline__ v16bf load_a_bf16(const __bf16* row, int k0, int half) {
  v16bf a;
  const __bf16* p = row + k0;
#pragma unroll
  for (int v = 0; v < 8; ++v) {
    int ko = (v < 4 ? 2 * v : 2 * v + 8) + 8 * half;
    a[2 * v]     = p[ko];
    a[2 * v + 1] = p[ko + 1];
  }
  return a;
}

// B-fragment for B = W^T with W row-major (N,K): lane L -> col N = L%16,
// VGPR v holds K-pair at offset (v<4 ? 2v : 2v+8) + 8*(L/16) (same as A).
__device__ __forceinline__ v16bf load_b_from_wT(const __bf16* wrow, int k0, int half) {
  v16bf b;
  const __bf16* p = wrow + k0;
#pragma unroll
  for (int v = 0; v < 8; ++v) {
    int ko = (v < 4 ? 2 * v : 2 * v + 8) + 8 * half;
    b[2 * v]     = p[ko];
    b[2 * v + 1] = p[ko + 1];
  }
  return b;
}

// ---------------- gfx1250 async global->LDS copy (guarded) ----------------
#if __has_builtin(__builtin_amdgcn_global_load_async_to_lds_b128)
#define HAVE_ASYNC_COPY 1
typedef __attribute__((address_space(1))) v4i_* as1_v4i_ptr;
typedef __attribute__((address_space(3))) v4i_* as3_v4i_ptr;
#endif

__device__ __forceinline__ void lds_copy16(__bf16* dst_lds, const __bf16* src_glob) {
#ifdef HAVE_ASYNC_COPY
  __builtin_amdgcn_global_load_async_to_lds_b128(
      (as1_v4i_ptr)(src_glob),
      (as3_v4i_ptr)(dst_lds),
      /*offset=*/0, /*cpol=*/0);
#else
  *reinterpret_cast<int4*>(dst_lds) = *reinterpret_cast<const int4*>(src_glob);
#endif
}

__device__ __forceinline__ void lds_copy_wait() {
#ifdef HAVE_ASYNC_COPY
#if __has_builtin(__builtin_amdgcn_s_wait_asynccnt)
  __builtin_amdgcn_s_wait_asynccnt(0);
#else
  asm volatile("s_wait_asynccnt 0" ::: "memory");
#endif
#endif
}

// ---------------- kernels ----------------

__global__ void cvt_f32_bf16_kernel(const float* __restrict__ src,
                                    __bf16* __restrict__ dst, int n) {
  int i = blockIdx.x * blockDim.x + threadIdx.x;
  int stride = gridDim.x * blockDim.x;
  for (; i < n; i += stride) dst[i] = (__bf16)src[i];
}

// Z[t*BATCH+b][n] = emb[X[b][t]] . W_ih[n] + b_ih[n] + b_hh[n]
__global__ void embed_proj_kernel(const int* __restrict__ X,
                                  const float* __restrict__ emb,
                                  const __bf16* __restrict__ Wih,
                                  const float* __restrict__ b_ih,
                                  const float* __restrict__ b_hh,
                                  float* __restrict__ Z) {
  const int lane = threadIdx.x & 31;
  const int wave = threadIdx.x >> 5;
  const int half = lane >> 4;
  const int m0 = blockIdx.x * 16;                 // row tile in [0, SEQ*BATCH)
  const int n0 = (blockIdx.y * 4 + wave) * 16;    // col tile in [0, HID)

  const int r = m0 + (lane & 15);                 // row r = t*BATCH + b
  const int t = r / BATCH;
  const int b = r % BATCH;
  const float* arow = emb + (size_t)X[b * SEQ + t] * EMB;
  const __bf16* brow = Wih + (size_t)(n0 + (lane & 15)) * EMB;

  v8f acc = {};
  for (int k0 = 0; k0 < EMB; k0 += 32) {
    v16bf a;
#pragma unroll
    for (int v = 0; v < 8; ++v) {               // fp32 emb -> bf16 on load
      int ko = (v < 4 ? 2 * v : 2 * v + 8) + 8 * half;
      a[2 * v]     = (__bf16)arow[k0 + ko];
      a[2 * v + 1] = (__bf16)arow[k0 + ko + 1];
    }
    v16bf bm = load_b_from_wT(brow, k0, half);
    acc = __builtin_amdgcn_wmma_f32_16x16x32_bf16(false, a, false, bm,
                                                  (short)0, acc, false, false);
  }
#pragma unroll
  for (int v = 0; v < 8; ++v) {
    int row = m0 + v + 8 * half;
    int col = n0 + (lane & 15);
    Z[(size_t)row * HID + col] = acc[v] + b_ih[col] + b_hh[col];
  }
}

// Persistent recurrence: one block per 16 batch rows runs ALL 512 steps.
// 32 waves, each owns one 16-col N-tile. h lives in LDS between steps.
// W_hh rows [0, RES_ROWS) pinned in LDS; rest streamed from L2.
constexpr int RES_ROWS = 256;                    // 256*512 bf16 = 256 KB in LDS
constexpr int RNN_LDS_BYTES = (RES_ROWS * HID + 16 * HID) * 2;  // 272 KB

__global__ __launch_bounds__(1024) void rnn_persistent_kernel(
    const float* __restrict__ Z,
    const __bf16* __restrict__ Whh,
    __bf16* __restrict__ Hout) {
  extern __shared__ __bf16 smem[];
  __bf16* Wres = smem;                      // [RES_ROWS][HID]
  __bf16* Abuf = smem + (size_t)RES_ROWS * HID;  // [16][HID] recurrent state

  const int tid  = threadIdx.x;
  const int lane = tid & 31;
  const int wave = tid >> 5;                // 0..31 -> n-tile
  const int half = lane >> 4;
  const int m0   = blockIdx.x * 16;         // batch rows owned by this block

  // ---- prologue: async-stage resident half of W_hh into LDS ----
  for (int idx = tid * 8; idx < RES_ROWS * HID; idx += 1024 * 8)
    lds_copy16(Wres + idx, Whh + idx);
  // zero the recurrent state buffer (h_{-1} = 0)
  for (int idx = tid; idx < 16 * HID; idx += 1024)
    Abuf[idx] = (__bf16)0.f;
  lds_copy_wait();
  __syncthreads();

  const int n0 = wave * 16;
  const bool resident = (n0 + 16) <= RES_ROWS;
  const __bf16* browL = Wres + (size_t)(n0 + (lane & 15)) * HID;
  const __bf16* browG = Whh + (size_t)(n0 + (lane & 15)) * HID;
  const __bf16* arow  = Abuf + (size_t)(lane & 15) * HID;

  for (int t = 0; t < SEQ; ++t) {
    v8f acc = {};
    if (t > 0) {
      if (resident) {
#pragma unroll 4
        for (int k0 = 0; k0 < HID; k0 += 32) {
          v16bf a  = load_a_bf16(arow, k0, half);   // ds_load
          v16bf bm = load_b_from_wT(browL, k0, half);
          acc = __builtin_amdgcn_wmma_f32_16x16x32_bf16(false, a, false, bm,
                                                        (short)0, acc, false, false);
        }
      } else {
#pragma unroll 4
        for (int k0 = 0; k0 < HID; k0 += 32) {
          __builtin_prefetch(browG + k0 + 128, 0, 1);
          v16bf a  = load_a_bf16(arow, k0, half);
          v16bf bm = load_b_from_wT(browG, k0, half);
          acc = __builtin_amdgcn_wmma_f32_16x16x32_bf16(false, a, false, bm,
                                                        (short)0, acc, false, false);
        }
      }
    }
    const float* Zt = Z + ((size_t)t * BATCH + m0) * HID;
    __syncthreads();                        // all reads of Abuf (h_{t-1}) done
#pragma unroll
    for (int v = 0; v < 8; ++v) {
      int row = v + 8 * half;               // local batch row
      int col = n0 + (lane & 15);
      __bf16 hb = (__bf16)tanhf(acc[v] + Zt[(size_t)row * HID + col]);
      Abuf[(size_t)row * HID + col] = hb;   // feeds next step's A-fragment
      Hout[((size_t)t * BATCH + m0 + row) * HID + col] = hb;
    }
    __syncthreads();                        // h_t visible in LDS
  }
}

// scores[t][b] = prev[t,b,:] . last[b,:]
__global__ void attn_scores_kernel(const __bf16* __restrict__ Hout,
                                   float* __restrict__ scores) {
  const int t = blockIdx.x;        // 0 .. SEQ-2
  const int b = threadIdx.x;       // 0 .. BATCH-1
  const __bf16* prev = Hout + ((size_t)t * BATCH + b) * HID;
  const __bf16* last = Hout + ((size_t)(SEQ - 1) * BATCH + b) * HID;
  float s = 0.f;
#pragma unroll 8
  for (int h = 0; h < HID; ++h) s += (float)prev[h] * (float)last[h];
  scores[t * BATCH + b] = s;
}

// per-batch softmax over t, weighted sum over prev, build concat [att_out, last]
__global__ void softmax_attout_kernel(const __bf16* __restrict__ Hout,
                                      const float* __restrict__ scores,
                                      __bf16* __restrict__ cat) {
  __shared__ float satt[SEQ];
  __shared__ float red[256];
  const int b = blockIdx.x;
  const int tid = threadIdx.x;
  const int T = SEQ - 1;

  float mx = -1e30f;
  for (int t = tid; t < T; t += 256) {
    float s = scores[t * BATCH + b];
    satt[t] = s;
    mx = fmaxf(mx, s);
  }
  red[tid] = mx;
  __syncthreads();
  for (int s = 128; s > 0; s >>= 1) {
    if (tid < s) red[tid] = fmaxf(red[tid], red[tid + s]);
    __syncthreads();
  }
  mx = red[0];
  __syncthreads();

  float sm = 0.f;
  for (int t = tid; t < T; t += 256) {
    float e = expf(satt[t] - mx);
    satt[t] = e;
    sm += e;
  }
  red[tid] = sm;
  __syncthreads();
  for (int s = 128; s > 0; s >>= 1) {
    if (tid < s) red[tid] += red[tid + s];
    __syncthreads();
  }
  const float inv = 1.f / red[0];
  __syncthreads();

  const __bf16* last = Hout + ((size_t)(SEQ - 1) * BATCH + b) * HID;
  for (int h = tid; h < HID; h += 256) {
    float a = 0.f;
    for (int t = 0; t < T; ++t)
      a += satt[t] * (float)Hout[((size_t)t * BATCH + b) * HID + h];
    cat[(size_t)b * CAT + h]       = (__bf16)(a * inv);
    cat[(size_t)b * CAT + HID + h] = last[h];
  }
}

// out = cat @ W^T + bias  (M=128, N=32000, K=1024).
// Each wave owns one N-tile and ALL 8 M-tiles (8 accumulators) so every
// B fragment of the 65 MB weight stream is loaded exactly once.
__global__ void classifier_kernel(const __bf16* __restrict__ cat,
                                  const __bf16* __restrict__ Wc,
                                  const float* __restrict__ bias,
                                  float* __restrict__ out) {
  const int lane = threadIdx.x & 31;
  const int wave = threadIdx.x >> 5;
  const int half = lane >> 4;
  const int n0 = (blockIdx.x * 4 + wave) * 16;

  const __bf16* brow = Wc + (size_t)(n0 + (lane & 15)) * CAT;
  const __bf16* abase = cat + (size_t)(lane & 15) * CAT;

  v8f acc[8] = {};
  for (int k0 = 0; k0 < CAT; k0 += 32) {
    __builtin_prefetch(brow + k0 + 256, 0, 1);   // global_prefetch_b8 on W stream
    v16bf bm = load_b_from_wT(brow, k0, half);
#pragma unroll
    for (int m = 0; m < 8; ++m) {
      v16bf a = load_a_bf16(abase + (size_t)m * 16 * CAT, k0, half);
      acc[m] = __builtin_amdgcn_wmma_f32_16x16x32_bf16(false, a, false, bm,
                                                       (short)0, acc[m], false, false);
    }
  }
  const int col = n0 + (lane & 15);
  const float bv = bias[col];
#pragma unroll
  for (int m = 0; m < 8; ++m)
#pragma unroll
    for (int v = 0; v < 8; ++v)
      out[(size_t)(m * 16 + v + 8 * half) * N_CLASS + col] = acc[m][v] + bv;
}

// ---------------- launch ----------------
extern "C" void kernel_launch(void* const* d_in, const int* in_sizes, int n_in,
                              void* d_out, int out_size, void* d_ws, size_t ws_size,
                              hipStream_t stream) {
  const int*   X    = (const int*)d_in[0];
  const float* emb  = (const float*)d_in[1];
  const float* W_ih = (const float*)d_in[2];
  const float* W_hh = (const float*)d_in[3];
  const float* b_ih = (const float*)d_in[4];
  const float* b_hh = (const float*)d_in[5];
  const float* W    = (const float*)d_in[6];
  const float* bias = (const float*)d_in[7];
  float* out = (float*)d_out;

  // workspace carve-up
  char* p = (char*)d_ws;
  __bf16* Wih_bf  = (__bf16*)p; p += (size_t)HID * EMB * 2;
  __bf16* Whh_bf  = (__bf16*)p; p += (size_t)HID * HID * 2;
  __bf16* Wcls_bf = (__bf16*)p; p += (size_t)N_CLASS * CAT * 2;
  float*  Z       = (float*)p;  p += (size_t)SEQ * BATCH * HID * 4;
  __bf16* Hout    = (__bf16*)p; p += (size_t)SEQ * BATCH * HID * 2;
  float*  scores  = (float*)p;  p += (size_t)SEQ * BATCH * 4;
  __bf16* cat     = (__bf16*)p; p += (size_t)BATCH * CAT * 2;

  // 1) weight conversion fp32 -> bf16
  cvt_f32_bf16_kernel<<<512, 256, 0, stream>>>(W_ih, Wih_bf, HID * EMB);
  cvt_f32_bf16_kernel<<<512, 256, 0, stream>>>(W_hh, Whh_bf, HID * HID);
  cvt_f32_bf16_kernel<<<2048, 256, 0, stream>>>(W, Wcls_bf, N_CLASS * CAT);

  // 2) hoisted input projection: Z = emb[X] @ W_ih^T + b_ih + b_hh
  embed_proj_kernel<<<dim3(SEQ * BATCH / 16, HID / 64), 128, 0, stream>>>(
      X, emb, Wih_bf, b_ih, b_hh, Z);

  // 3) full recurrence in ONE launch: 8 persistent blocks (one per 16 batch
  //    rows), h kept in LDS, s_barrier between steps, W_hh half LDS-resident.
  rnn_persistent_kernel<<<BATCH / 16, 1024, RNN_LDS_BYTES, stream>>>(Z, Whh_bf, Hout);

  // 4) attention
  attn_scores_kernel<<<SEQ - 1, BATCH, 0, stream>>>(Hout, scores);
  softmax_attout_kernel<<<BATCH, 256, 0, stream>>>(Hout, scores, cat);

  // 5) classifier GEMM: W streamed exactly once
  classifier_kernel<<<N_CLASS / 64, 128, 0, stream>>>(cat, Wcls_bf, bias, out);
}
